// LiGRULayer_30159260353160
// MI455X (gfx1250) — compile-verified, matrix operand order checked by reading, not run on hardware
//
#include <hip/hip_runtime.h>
#include <hip/hip_bf16.h>

typedef __bf16 bf16_t;
typedef __attribute__((ext_vector_type(16))) __bf16 bf16x16;
typedef __attribute__((ext_vector_type(8)))  float  f32x8;
typedef __attribute__((ext_vector_type(4)))  float  f32x4;
typedef __attribute__((ext_vector_type(4)))  int    v4i;

#define B_   32
#define T_   1000
#define I_   512
#define H_   1024
#define M_   (B_ * T_)     // 32000 rows
#define EPS_ 1e-5f

#define NBLK 64            // persistent blocks in recurrence (H_/COLS)
#define COLS 16            // output columns per block
#define VPAD 8             // bf16 pad per LDS row -> +16B, kills bank conflicts

union LdU { f32x4 q[2]; bf16x16 v; };

// Load one wave-lane's A/B fragment for v_wmma_f32_16x16x32_bf16.
// Per the CDNA5 16-bit A layout, lane (row = lane%16, half = lane/16) holds
// two contiguous 8-element K-chunks at k0 + 8*half and k0 + 16 + 8*half.
__device__ __forceinline__ bf16x16 ld_frag(const bf16_t* p) {
  LdU u;
  u.q[0] = *reinterpret_cast<const f32x4*>(p);
  u.q[1] = *reinterpret_cast<const f32x4*>(p + 16);
  return u.v;
}

// ---- CDNA5 async global->LDS (ASYNCcnt) with portable fallbacks ----------
__device__ __forceinline__ void async_ld_b128(const bf16_t* g, bf16_t* l) {
#if __has_builtin(__builtin_amdgcn_global_load_async_to_lds_b128)
  __builtin_amdgcn_global_load_async_to_lds_b128(
      (__attribute__((address_space(1))) v4i*)g,
      (__attribute__((address_space(3))) v4i*)l, 0, 0);
#else
  *reinterpret_cast<f32x4*>(l) = *reinterpret_cast<const f32x4*>(g);
#endif
}

__device__ __forceinline__ void wait_async0() {
#if __has_builtin(__builtin_amdgcn_s_wait_asynccnt)
  __builtin_amdgcn_s_wait_asynccnt(0);
#else
  asm volatile("s_wait_asynccnt 0" ::: "memory");
#endif
}

// ---------------------------------------------------------------- convert
__global__ void k_f32_to_bf16(const float* __restrict__ src,
                              bf16_t* __restrict__ dst, int n) {
  int i = blockIdx.x * blockDim.x + threadIdx.x;
  int stride = gridDim.x * blockDim.x;
  for (; i < n; i += stride) dst[i] = (bf16_t)src[i];
}

// ---------------------------------------------------------------- FF GEMM
// C[M_ x H_] = Xb[M_ x I_] * Wb[H_ x I_]^T  (TN), both gates via blockIdx.z.
// Block = 8 waves (256 thr), tile 128M x 128N; wave tile 32M x 64N.
__global__ __launch_bounds__(256)
void k_ff_gemm(const bf16_t* __restrict__ xb,
               const bf16_t* __restrict__ wb,
               const bf16_t* __restrict__ wzb,
               float* __restrict__ wx, float* __restrict__ wzx) {
  const int gate = blockIdx.z;
  const bf16_t* w = gate ? wzb : wb;
  float* outp     = gate ? wzx : wx;

  const int tid  = threadIdx.x;
  const int lane = tid & 31;
  const int wave = tid >> 5;
  const int wm   = wave & 3;     // 0..3 -> M sub-tile
  const int wn   = wave >> 2;    // 0..1 -> N sub-tile
  const int Mb   = blockIdx.y * 128;
  const int Nb   = blockIdx.x * 128;
  const int lr   = lane & 15;
  const int lg8  = (lane >> 4) * 8;

  f32x8 acc[2][4];
  f32x8 zero = {};
  for (int mt = 0; mt < 2; ++mt)
    for (int nt = 0; nt < 4; ++nt) acc[mt][nt] = zero;

  const bf16_t* a0 = xb + (size_t)(Mb + wm * 32 + lr) * I_ + lg8;
  const bf16_t* a1 = a0 + (size_t)16 * I_;
  const bf16_t* bptr[4];
  for (int nt = 0; nt < 4; ++nt)
    bptr[nt] = w + (size_t)(Nb + wn * 64 + nt * 16 + lr) * I_ + lg8;

  for (int k0 = 0; k0 < I_; k0 += 32) {
    bf16x16 av0 = ld_frag(a0 + k0);
    bf16x16 av1 = ld_frag(a1 + k0);
#pragma unroll
    for (int nt = 0; nt < 4; ++nt) {
      bf16x16 bv = ld_frag(bptr[nt] + k0);
      acc[0][nt] = __builtin_amdgcn_wmma_f32_16x16x32_bf16(
          false, av0, false, bv, (short)0, acc[0][nt], false, false);
      acc[1][nt] = __builtin_amdgcn_wmma_f32_16x16x32_bf16(
          false, av1, false, bv, (short)0, acc[1][nt], false, false);
    }
  }

  // C/D layout: VGPR j holds row (j + 8*(lane/16)), column lane%16
#pragma unroll
  for (int mt = 0; mt < 2; ++mt)
#pragma unroll
    for (int nt = 0; nt < 4; ++nt) {
      int n = Nb + wn * 64 + nt * 16 + lr;
#pragma unroll
      for (int j = 0; j < 8; ++j) {
        int m = Mb + wm * 32 + mt * 16 + j + lg8;
        outp[(size_t)m * H_ + n] = acc[mt][nt][j];
      }
    }
}

// ---------------------------------------------------------------- BN stats
// Stage 1: partial sum / sumsq over 256-row chunks, fully coalesced, no atomics.
__global__ __launch_bounds__(256)
void k_bn_partial(const float* __restrict__ wx, const float* __restrict__ wzx,
                  float* __restrict__ psum, float* __restrict__ psqs) {
  const int g = blockIdx.z;
  const float* src = g ? wzx : wx;
  const int c  = blockIdx.x * 256 + threadIdx.x;
  const int r0 = blockIdx.y * 256;
  float s = 0.f, s2 = 0.f;
  for (int r = 0; r < 256; ++r) {
    float v = src[(size_t)(r0 + r) * H_ + c];
    s += v; s2 += v * v;
  }
  int idx = (g * 125 + blockIdx.y) * H_ + c;
  psum[idx] = s;
  psqs[idx] = s2;
}

// Stage 2: finalize mean/var -> affine scale/shift per (gate, column)
__global__ void k_bn_finalize(const float* __restrict__ psum,
                              const float* __restrict__ psqs,
                              const float* __restrict__ gamma,
                              const float* __restrict__ beta,
                              const float* __restrict__ gammaz,
                              const float* __restrict__ betaz,
                              float* __restrict__ bnsc, float* __restrict__ bnsh) {
  int idx = blockIdx.x * blockDim.x + threadIdx.x;
  if (idx >= 2 * H_) return;
  int g = idx >> 10, c = idx & (H_ - 1);
  float s = 0.f, s2 = 0.f;
  for (int i = 0; i < 125; ++i) {
    s  += psum[(g * 125 + i) * H_ + c];
    s2 += psqs[(g * 125 + i) * H_ + c];
  }
  float inv  = 1.f / (float)M_;
  float mean = s * inv;
  float var  = s2 * inv - mean * mean;
  float ga   = g ? gammaz[c] : gamma[c];
  float be   = g ? betaz[c]  : beta[c];
  float sc   = ga * rsqrtf(var + EPS_);
  bnsc[idx] = sc;
  bnsh[idx] = be - mean * sc;
}

// ---------------------------------------------------------------- init
__global__ void k_init(unsigned* __restrict__ ybf0_u32, unsigned* __restrict__ bar) {
  int i = blockIdx.x * blockDim.x + threadIdx.x;
  if (i < (B_ * H_ * 2) / 4) ybf0_u32[i] = 0u;  // y0 = 0 (bf16 buffer)
  if (i == 0) *bar = 0u;                        // grid barrier counter
}

// ---------------------------------------------------------------- recurrence
// Persistent kernel: NBLK blocks x 128 threads (4 waves). Block owns H-slice
// [c0, c0+16). V/Vz slices live in LDS; y_prev (bf16, 64KB) is async-staged
// into LDS each step; one global barrier per timestep; y f32 state in LDS.
__global__ __launch_bounds__(128)
void k_recurrence(const float* __restrict__ wx, const float* __restrict__ wzx,
                  const float* __restrict__ bnsc, const float* __restrict__ bnsh,
                  const bf16_t* __restrict__ vb, const bf16_t* __restrict__ vzb,
                  bf16_t* __restrict__ ybf0, bf16_t* __restrict__ ybf1,
                  float* __restrict__ out, unsigned* __restrict__ bar) {
  __shared__ __align__(16) bf16_t sV[2][COLS][H_ + VPAD]; // ~66 KB weights
  __shared__ __align__(16) bf16_t sA[B_][H_ + VPAD];      // ~66 KB staged y
  __shared__ float sC[4][16][16];                         // per-wave 16x16 C
  __shared__ float sY[B_][COLS];                          // f32 y slice

  const int tid  = threadIdx.x;
  const int lane = tid & 31;
  const int wave = tid >> 5;      // 0..3
  const int gate = wave & 1;      // 0 -> V (candidate), 1 -> Vz (update gate)
  const int mt   = wave >> 1;     // 0..1 -> batch rows 0-15 / 16-31
  const int c0   = blockIdx.x * COLS;
  const int lr   = lane & 15;
  const int lg8  = (lane >> 4) * 8;

  // Preload recurrent weight slices: sV[g][n][k] = Vg[(c0+n)*H_ + k]
  for (int i = tid; i < 2 * COLS * (H_ / 8); i += 128) {
    int g   = i / (COLS * (H_ / 8));
    int rem = i % (COLS * (H_ / 8));
    int n   = rem / (H_ / 8);
    int kc  = rem % (H_ / 8);
    const bf16_t* src = (g ? vzb : vb) + (size_t)(c0 + n) * H_ + kc * 8;
    *reinterpret_cast<f32x4*>(&sV[g][n][kc * 8]) =
        *reinterpret_cast<const f32x4*>(src);
  }
  for (int i = tid; i < B_ * COLS; i += 128) sY[i / COLS][i % COLS] = 0.f;

  // Hoist per-thread epilogue constants (each thread owns 4 fixed (m,n) cells)
  int em[4], en[4];
  float sc_c[4], sh_c[4], sc_z[4], sh_z[4];
#pragma unroll
  for (int i = 0; i < 4; ++i) {
    int e = tid + 128 * i;
    em[i] = e >> 4;
    en[i] = e & 15;
    int c = c0 + en[i];
    sc_c[i] = bnsc[c];        sh_c[i] = bnsh[c];
    sc_z[i] = bnsc[H_ + c];   sh_z[i] = bnsh[H_ + c];
  }
  __syncthreads();

  unsigned target = 0;
  for (int t = 0; t < T_; ++t) {
    // ---- grid-wide barrier: y_bf16 of step t-1 visible everywhere ----
    __threadfence();
    __syncthreads();
    if (tid == 0) {
      atomicAdd(bar, 1u);
      target += NBLK;
      while (__hip_atomic_load(bar, __ATOMIC_ACQUIRE,
                               __HIP_MEMORY_SCOPE_AGENT) < target)
        __builtin_amdgcn_s_sleep(2);
    }
    __syncthreads();
    __threadfence();

    const bf16_t* yprev = (t & 1) ? ybf1 : ybf0;
    bf16_t*       ycur  = (t & 1) ? ybf0 : ybf1;

    // ---- async-stage y_prev [32 x 1024] bf16 into LDS (64 KB) ----
    // 4096 16-byte chunks, 128 threads -> 32 async b128 ops per thread.
    for (int i = tid; i < B_ * (H_ / 8); i += 128) {
      int m  = i >> 7;          // H_/8 = 128 chunks per row
      int kc = i & 127;
      async_ld_b128(yprev + (size_t)m * H_ + kc * 8, &sA[m][kc * 8]);
    }
    wait_async0();
    __syncthreads();

    // ---- wave GEMM: C(16x16) = y_prev[mt rows] @ Vg[c0..c0+15]^T ----
    f32x8 acc = {};
    const bf16_t* ap = &sA[mt * 16 + lr][lg8];
    const bf16_t* bp = &sV[gate][lr][lg8];
#pragma unroll 4
    for (int k0 = 0; k0 < H_; k0 += 32) {
      bf16x16 av = ld_frag(ap + k0);
      bf16x16 bv = ld_frag(bp + k0);
      acc = __builtin_amdgcn_wmma_f32_16x16x32_bf16(
          false, av, false, bv, (short)0, acc, false, false);
    }
#pragma unroll
    for (int j = 0; j < 8; ++j) sC[wave][j + lg8][lr] = acc[j];
    __syncthreads();

    // ---- elementwise LiGRU update + BN affine on streamed Wx/Wzx ----
#pragma unroll
    for (int i = 0; i < 4; ++i) {
      int m = em[i], n = en[i];
      int c = c0 + n;
      size_t row = (size_t)m * T_ + t;     // (batch, time) flat row
      float wxv  = fmaf(wx [row * H_ + c], sc_c[i], sh_c[i]);
      float wzxv = fmaf(wzx[row * H_ + c], sc_z[i], sh_z[i]);
      float cc = wxv  + sC[2 * (m >> 4) + 0][m & 15][n];
      float zz = wzxv + sC[2 * (m >> 4) + 1][m & 15][n];
      float z  = 1.f / (1.f + __expf(-zz));
      float ct = fmaxf(cc, 0.f);
      float yo = sY[m][n];
      float yn = z * yo + (1.f - z) * ct;
      sY[m][n] = yn;
      out[row * H_ + c]        = yn;
      ycur[(size_t)m * H_ + c] = (bf16_t)yn;
      // Prefetch next timestep's streamed BN inputs (read-once from HBM):
      if (t + 1 < T_) {
        __builtin_prefetch(&wx [(row + 1) * H_ + c], 0, 3);
        __builtin_prefetch(&wzx[(row + 1) * H_ + c], 0, 3);
      }
    }
    // sC/sA reuse protected by the barrier's __syncthreads at next loop top
  }
}

// ---------------------------------------------------------------- launch
extern "C" void kernel_launch(void* const* d_in, const int* in_sizes, int n_in,
                              void* d_out, int out_size, void* d_ws, size_t ws_size,
                              hipStream_t stream) {
  const float* x   = (const float*)d_in[0];
  const float* W   = (const float*)d_in[1];
  const float* Wz  = (const float*)d_in[2];
  const float* V   = (const float*)d_in[3];
  const float* Vz  = (const float*)d_in[4];
  const float* gm  = (const float*)d_in[5];
  const float* bt  = (const float*)d_in[6];
  const float* gmz = (const float*)d_in[7];
  const float* btz = (const float*)d_in[8];
  float* out = (float*)d_out;

  char* p = (char*)d_ws;
  auto take = [&](size_t bytes) -> char* {
    char* r = p;
    p += (bytes + 255) & ~(size_t)255;
    return r;
  };
  bf16_t* xb   = (bf16_t*)take((size_t)M_ * I_ * 2);
  bf16_t* wb   = (bf16_t*)take((size_t)H_ * I_ * 2);
  bf16_t* wzb  = (bf16_t*)take((size_t)H_ * I_ * 2);
  bf16_t* vbw  = (bf16_t*)take((size_t)H_ * H_ * 2);
  bf16_t* vzbw = (bf16_t*)take((size_t)H_ * H_ * 2);
  float*  wxf  = (float*)take((size_t)M_ * H_ * 4);
  float*  wzxf = (float*)take((size_t)M_ * H_ * 4);
  float*  psum = (float*)take((size_t)2 * 125 * H_ * 4);
  float*  psqs = (float*)take((size_t)2 * 125 * H_ * 4);
  float*  bnsc = (float*)take((size_t)2 * H_ * 4);
  float*  bnsh = (float*)take((size_t)2 * H_ * 4);
  bf16_t* ybf0 = (bf16_t*)take((size_t)B_ * H_ * 2);
  bf16_t* ybf1 = (bf16_t*)take((size_t)B_ * H_ * 2);
  unsigned* bar = (unsigned*)take(256);

  k_f32_to_bf16<<<4096, 256, 0, stream>>>(x,  xb,  M_ * I_);
  k_f32_to_bf16<<<2048, 256, 0, stream>>>(W,  wb,  H_ * I_);
  k_f32_to_bf16<<<2048, 256, 0, stream>>>(Wz, wzb, H_ * I_);
  k_f32_to_bf16<<<4096, 256, 0, stream>>>(V,  vbw, H_ * H_);
  k_f32_to_bf16<<<4096, 256, 0, stream>>>(Vz, vzbw, H_ * H_);

  k_ff_gemm<<<dim3(8, 250, 2), 256, 0, stream>>>(xb, wb, wzb, wxf, wzxf);

  k_bn_partial<<<dim3(4, 125, 2), 256, 0, stream>>>(wxf, wzxf, psum, psqs);
  k_bn_finalize<<<8, 256, 0, stream>>>(psum, psqs, gm, bt, gmz, btz, bnsc, bnsh);

  k_init<<<64, 256, 0, stream>>>((unsigned*)ybf0, bar);

  k_recurrence<<<NBLK, 128, 0, stream>>>(wxf, wzxf, bnsc, bnsh,
                                         vbw, vzbw, ybf0, ybf1, out, bar);
}
